// Decoder_47081431498884
// MI455X (gfx1250) — matrix operand order repaired and checked by reading, not compiled, not run on hardware
//
#include <hip/hip_runtime.h>
#include <stdint.h>

#define DEVI __device__ __forceinline__

typedef __bf16 bf16_t;
typedef __attribute__((ext_vector_type(16))) __bf16 v16bf;
typedef __attribute__((ext_vector_type(8)))  __bf16 bf16x8;
typedef __attribute__((ext_vector_type(8)))  float  v8f;
typedef __attribute__((ext_vector_type(4)))  unsigned int u32x4;
typedef __attribute__((ext_vector_type(8)))  int    i32x8;
typedef __attribute__((ext_vector_type(4)))  int    i32x4;

constexpr int B_   = 32;
constexpr int TDEC = 500;
constexpr int TENC = 256;
constexpr int D_   = 512;
constexpr int MEL_ = 80;
constexpr int OC_  = 32;
constexpr int KS_  = 31;
constexpr int NWG  = 16;    // persistent workgroups
constexpr int NROW = B_ * TDEC;   // 16000

// ---------------- scalar helpers ----------------
DEVI bf16_t f2bf(float f) {
  union { float f; unsigned u; } a; a.f = f;
  unsigned r = a.u + 0x7FFFu + ((a.u >> 16) & 1u);   // RNE
  union { unsigned short s; bf16_t b; } c; c.s = (unsigned short)(r >> 16);
  return c.b;
}
DEVI float sigm_(float x) { return 1.f / (1.f + __expf(-x)); }
DEVI float tanh_(float x) { return 1.f - 2.f / (__expf(2.f * x) + 1.f); }
DEVI unsigned hashu(unsigned x) {
  x ^= x >> 16; x *= 0x7feb352du; x ^= x >> 15; x *= 0x846ca68bu; x ^= x >> 16; return x;
}

// ---------------- WMMA fragment helpers (bf16 16x16x32) ----------------
// A (16x32, row major source [rows][K]): lane l -> row (l&15); elem i<8 -> k=koff+(l>>4)*8+i ; i>=8 -> +16
DEVI v16bf load_a16(const bf16_t* base, int stride, int row0, int koff) {
  const int l = threadIdx.x & 31;
  const bf16_t* p = base + (size_t)(row0 + (l & 15)) * stride + koff + (l >> 4) * 8;
  union { v16bf v; bf16x8 h[2]; } u;
  u.h[0] = *(const bf16x8*)(p);
  u.h[1] = *(const bf16x8*)(p + 16);
  return u.v;
}
// B (32x16 = K x N) from weights stored [out=N][in=K] row major: lane l -> col (l&15); elem i<8 -> k=koff+(l>>4)*8+i ; i>=8 -> +16
DEVI v16bf load_b16(const bf16_t* base, int stride, int col0, int koff) {
  const int l = threadIdx.x & 31;
  const bf16_t* p = base + (size_t)(col0 + (l & 15)) * stride + koff + (l >> 4) * 8;
  union { v16bf v; bf16x8 h[2]; } u;
  u.h[0] = *(const bf16x8*)(p);
  u.h[1] = *(const bf16x8*)(p + 16);
  return u.v;
}
DEVI v8f wmma_bf(v16bf a, v16bf b, v8f c) {
  return __builtin_amdgcn_wmma_f32_16x16x32_bf16(false, a, false, b, (short)0, c, false, false);
}

// ---------------- grid split-barrier (agent scope) ----------------
DEVI void grid_barrier(unsigned* bar, int nwg) {
  __syncthreads();
  if (threadIdx.x == 0) {
    unsigned* cnt = bar;
    unsigned* gen = bar + 1;
    unsigned g = __hip_atomic_load(gen, __ATOMIC_RELAXED, __HIP_MEMORY_SCOPE_AGENT);
    unsigned prev = __hip_atomic_fetch_add(cnt, 1u, __ATOMIC_ACQ_REL, __HIP_MEMORY_SCOPE_AGENT);
    if (prev == (unsigned)(nwg - 1)) {
      __hip_atomic_store(cnt, 0u, __ATOMIC_RELAXED, __HIP_MEMORY_SCOPE_AGENT);
      __hip_atomic_fetch_add(gen, 1u, __ATOMIC_ACQ_REL, __HIP_MEMORY_SCOPE_AGENT);
    } else {
      while (__hip_atomic_load(gen, __ATOMIC_ACQUIRE, __HIP_MEMORY_SCOPE_AGENT) == g)
        __builtin_amdgcn_s_sleep(4);
    }
  }
  __syncthreads();
}

// ---------------- Tensor Data Mover: 1D global->LDS copy ----------------
// D# built per CDNA5 ISA (group0: count/lds/global/type; group1: dims/strides)
DEVI void tdm_load_1d(void* lds_dst, const void* gsrc, unsigned bytes) {
  if (threadIdx.x == 0) {
    const unsigned ndw = bytes >> 2;                       // data_size = 4B units
    const unsigned lds = (unsigned)(uintptr_t)lds_dst;     // low 32 bits = LDS byte address
    const unsigned long long ga = (unsigned long long)(uintptr_t)gsrc;
    u32x4 g0;
    g0[0] = 1u;                                            // count=1, user descriptor
    g0[1] = lds;                                           // lds_addr
    g0[2] = (unsigned)ga;                                  // global_addr[31:0]
    g0[3] = (unsigned)((ga >> 32) & 0x1FFFFFFu) | (2u << 30); // global_addr[56:32] | type=2
    i32x8 g1;
    g1[0] = (int)(2u << 16);                               // data_size=4B, no flags
    g1[1] = (int)((ndw & 0xFFFFu) << 16);                  // tensor_dim0 lo16
    g1[2] = (int)((ndw >> 16) | (1u << 16));               // tensor_dim0 hi16 | tensor_dim1=1
    g1[3] = (int)((ndw & 0xFFFFu) << 16);                  // tile_dim0 = ndw
    g1[4] = 0;                                             // tile_dim1=0 (1D), tile_dim2=0
    g1[5] = (int)ndw;                                      // tensor_dim0_stride lo32
    g1[6] = 0;
    g1[7] = 0;
    i32x4 z4 = {0, 0, 0, 0};
#if defined(__clang_major__) && (__clang_major__ >= 23)
    i32x8 z8 = {0, 0, 0, 0, 0, 0, 0, 0};
    __builtin_amdgcn_tensor_load_to_lds(g0, g1, z4, z4, z8, 0);
#else
    __builtin_amdgcn_tensor_load_to_lds(g0, g1, z4, z4, 0);
#endif
    __builtin_amdgcn_s_wait_tensorcnt(0);
  }
}

// =======================================================================
// Pre-pass kernels
// =======================================================================
__global__ void k_pack_plain(const float* src, bf16_t* dst, int n) {
  int i = blockIdx.x * 256 + threadIdx.x;
  if (i < n) dst[i] = f2bf(src[i]);
}
// permute LSTM weight rows: packed r = 4*j+g  <-  orig g*512+j  (gate interleave)
__global__ void k_pack_lstm_w(const float* src, bf16_t* dst, int K) {
  int i = blockIdx.x * 256 + threadIdx.x;
  if (i < 2048 * K) {
    int r = i / K, k = i - r * K;
    int orig = (r & 3) * 512 + (r >> 2);
    dst[i] = f2bf(src[orig * K + k]);
  }
}
__global__ void k_pack_lstm_b(const float* bih, const float* bhh, float* dst) {
  int r = blockIdx.x * 256 + threadIdx.x;
  if (r < 2048) {
    int orig = (r & 3) * 512 + (r >> 2);
    dst[r] = bih[orig] + bhh[orig];
  }
}

// prenet layer 1 (K=80, VALU) -> h1 bf16 [16000][256]
__global__ void k_prenet1(const float* mel_in, const float* w1, const float* b1, bf16_t* h1) {
  int r = blockIdx.x, c = threadIdx.x;
  const float* x = mel_in + (size_t)r * MEL_;
  const float* w = w1 + (size_t)c * MEL_;
  float s = b1[c];
  #pragma unroll 8
  for (int k = 0; k < MEL_; ++k) s += x[k] * w[k];
  s = fmaxf(s, 0.f);
  unsigned h = hashu((unsigned)(r * 256 + c) ^ 0x9E3779B9u);
  s = (h & 1u) ? s * 2.f : 0.f;
  h1[(size_t)r * 256 + c] = f2bf(s);
}

// prenet layer 2 (WMMA, M=16000 N=256 K=256) -> pm bf16
__global__ __launch_bounds__(256) void k_prenet2(const bf16_t* h1, const bf16_t* w2bf,
                                                 const float* b2, bf16_t* pm) {
  int gw = blockIdx.x * 8 + (threadIdx.x >> 5);   // wave id = tile id (16000 tiles)
  int mt = gw >> 4, nt = gw & 15;
  int lane = threadIdx.x & 31;
  v8f c = {0.f, 0.f, 0.f, 0.f, 0.f, 0.f, 0.f, 0.f};
  for (int k = 0; k < 256; k += 32) {
    v16bf a = load_a16(h1, 256, mt * 16, k);
    v16bf b = load_b16(w2bf, 256, nt * 16, k);
    c = wmma_bf(a, b, c);
  }
  int half = lane >> 4, nl = lane & 15;
  #pragma unroll
  for (int v = 0; v < 8; ++v) {
    int r = mt * 16 + v + 8 * half;
    int cc = nt * 16 + nl;
    float s = fmaxf(c[v] + b2[cc], 0.f);
    unsigned h = hashu((unsigned)(r * 256 + cc) ^ 0x85EBCA6Bu);
    s = (h & 1u) ? s * 2.f : 0.f;
    pm[(size_t)r * 256 + cc] = f2bf(s);
  }
}

// m_proj = enc @ mem_w^T + mem_b  (WMMA, M=8192 N=512 K=512) -> f32
__global__ __launch_bounds__(256) void k_mproj(const bf16_t* encbf, const bf16_t* wbf,
                                               const float* mb, float* mproj) {
  int gw = blockIdx.x * 8 + (threadIdx.x >> 5);   // 16384 tiles
  int mt = gw >> 5, nt = gw & 31;
  int lane = threadIdx.x & 31;
  v8f c = {0.f, 0.f, 0.f, 0.f, 0.f, 0.f, 0.f, 0.f};
  for (int k = 0; k < 512; k += 32) {
    v16bf a = load_a16(encbf, 512, mt * 16, k);
    v16bf b = load_b16(wbf, 512, nt * 16, k);
    c = wmma_bf(a, b, c);
  }
  int half = lane >> 4, nl = lane & 15;
  #pragma unroll
  for (int v = 0; v < 8; ++v) {
    int r = mt * 16 + v + 8 * half;
    int cc = nt * 16 + nl;
    mproj[(size_t)r * 512 + cc] = c[v] + mb[cc];
  }
}

// =======================================================================
// Persistent decode kernel (16 WGs x 256 threads, 500 sequential steps)
// =======================================================================
struct DecP {
  const float *enc, *conv_w, *conv_b, *locp_b, *v_w, *v_b, *q_b, *mel_w, *mel_b, *stop_w, *stop_b;
  const bf16_t *att_wih, *att_whh, *dec_wih, *dec_whh, *q_wbf, *locp_wbf, *pm;
  const float *att_b, *dec_b, *mproj;
  float *q, *e, *a, *acum, *ctx, *c_att, *c_dec, *h_dec;
  bf16_t *conv_out, *h_att_bf, *h_dec_bf, *xin_att, *xin_dec;
  float *out_mel, *out_stop;
  unsigned* bar;
};

__global__ __launch_bounds__(256) void k_decode(DecP p) {
  __shared__ bf16_t s_xin[B_ * 768];     // 48 KB : TDM-staged LSTM input matrix
  __shared__ float  s_gates[B_ * 128];   // 16 KB : gate transpose buffer

  const int tid  = threadIdx.x;
  const int wg   = blockIdx.x;          // 0..15
  const int wave = tid >> 5;
  const int lane = tid & 31;
  const int gtid = wg * 256 + tid;
  const int half = lane >> 4, nl = lane & 15;

  // ---- init state (poisoned ws -> must zero everything we read) ----
  for (int i = gtid; i < B_ * TENC; i += NWG * 256) { p.a[i] = 0.f; p.acum[i] = 0.f; }
  for (int i = gtid; i < B_ * D_; i += NWG * 256) {
    p.c_att[i] = 0.f; p.c_dec[i] = 0.f; p.ctx[i] = 0.f; p.h_dec[i] = 0.f;
  }
  for (int i = gtid; i < 2 * B_ * D_; i += NWG * 256) {
    p.h_att_bf[i] = f2bf(0.f); p.h_dec_bf[i] = f2bf(0.f);
  }
  for (int i = gtid; i < B_ * 768; i += NWG * 256) {
    int b = i / 768, k = i - b * 768;
    p.xin_att[i] = (k < 256) ? p.pm[((size_t)b * TDEC + 0) * 256 + k] : f2bf(0.f);
  }
  for (int i = gtid; i < B_ * 1024; i += NWG * 256) p.xin_dec[i] = f2bf(0.f);
  grid_barrier(p.bar, NWG);

  for (int t = 0; t < TDEC; ++t) {
    const int prev = t & 1, cur = (t + 1) & 1;

    // ============ S1: attention LSTM (WG owns 32 hidden units = 128 gate cols) ============
    tdm_load_1d(s_xin, p.xin_att, B_ * 768 * (unsigned)sizeof(bf16_t));
    __syncthreads();
    {
      const int t0 = wave * 2;                 // two C tiles per wave
      const int mi = t0 >> 3;
      const int n0 = (t0 & 7) * 16, n1 = n0 + 16;
      const int colbase = wg * 128;
      v8f c0 = {0.f,0.f,0.f,0.f,0.f,0.f,0.f,0.f};
      v8f c1 = {0.f,0.f,0.f,0.f,0.f,0.f,0.f,0.f};
      for (int k = 0; k < 768; k += 32) {      // input part (pm || ctx) from LDS
        v16bf af = load_a16(s_xin, 768, mi * 16, k);
        v16bf b0 = load_b16(p.att_wih, 768, colbase + n0, k);
        v16bf b1 = load_b16(p.att_wih, 768, colbase + n1, k);
        c0 = wmma_bf(af, b0, c0);
        c1 = wmma_bf(af, b1, c1);
      }
      const bf16_t* hp = p.h_att_bf + prev * B_ * D_;
      for (int k = 0; k < 512; k += 32) {      // recurrent part
        v16bf af = load_a16(hp, 512, mi * 16, k);
        v16bf b0 = load_b16(p.att_whh, 512, colbase + n0, k);
        v16bf b1 = load_b16(p.att_whh, 512, colbase + n1, k);
        c0 = wmma_bf(af, b0, c0);
        c1 = wmma_bf(af, b1, c1);
      }
      #pragma unroll
      for (int v = 0; v < 8; ++v) {
        int b = mi * 16 + v + 8 * half;
        s_gates[b * 128 + n0 + nl] = c0[v];
        s_gates[b * 128 + n1 + nl] = c1[v];
      }
    }
    __syncthreads();
    for (int idx = tid; idx < B_ * 32; idx += 256) {
      int b = idx >> 5, jl = idx & 31;
      int col = wg * 128 + jl * 4;
      float gi = s_gates[b * 128 + jl * 4 + 0] + p.att_b[col + 0];
      float gf = s_gates[b * 128 + jl * 4 + 1] + p.att_b[col + 1];
      float gg = s_gates[b * 128 + jl * 4 + 2] + p.att_b[col + 2];
      float go = s_gates[b * 128 + jl * 4 + 3] + p.att_b[col + 3];
      int j = wg * 32 + jl;
      float cn = sigm_(gf) * p.c_att[b * D_ + j] + sigm_(gi) * tanh_(gg);
      float h  = sigm_(go) * tanh_(cn);
      p.c_att[b * D_ + j] = cn;
      p.h_att_bf[cur * B_ * D_ + b * D_ + j] = f2bf(h);
      p.xin_dec[b * 1024 + j] = f2bf(h);
    }
    grid_barrier(p.bar, NWG);

    // ============ S2: query projection (waves 0..3) + location conv (all) ============
    if (wave < 4) {
      int tile = wg * 4 + wave;                 // 64 tiles (2 x 32)
      int mi = tile >> 5, nt = tile & 31;
      const bf16_t* hc = p.h_att_bf + cur * B_ * D_;
      v8f c = {0.f,0.f,0.f,0.f,0.f,0.f,0.f,0.f};
      for (int k = 0; k < 512; k += 32) {
        v16bf af = load_a16(hc, 512, mi * 16, k);
        v16bf bf = load_b16(p.q_wbf, 512, nt * 16, k);
        c = wmma_bf(af, bf, c);
      }
      #pragma unroll
      for (int v = 0; v < 8; ++v) {
        int b = mi * 16 + v + 8 * half;
        int d = nt * 16 + nl;
        p.q[b * D_ + d] = c[v] + p.q_b[d];
      }
    }
    for (int pr = gtid * 2; pr < gtid * 2 + 2; ++pr) {   // 8192 (b,t) pairs
      int b = pr >> 8, tt = pr & 255;
      float wa[KS_], wc[KS_];
      #pragma unroll
      for (int k = 0; k < KS_; ++k) {
        int tp = tt + k - KS_ / 2;
        bool ok = (tp >= 0) && (tp < TENC);
        wa[k] = ok ? p.a[b * TENC + tp] : 0.f;
        wc[k] = ok ? p.acum[b * TENC + tp] : 0.f;
      }
      for (int oc = 0; oc < OC_; ++oc) {
        const float* w0 = p.conv_w + (oc * 2 + 0) * KS_;
        const float* w1 = p.conv_w + (oc * 2 + 1) * KS_;
        float s = p.conv_b[oc];
        #pragma unroll
        for (int k = 0; k < KS_; ++k) s += wa[k] * w0[k] + wc[k] * w1[k];
        p.conv_out[(size_t)pr * OC_ + oc] = f2bf(s);     // conv bias folded in pre-proj
      }
    }
    grid_barrier(p.bar, NWG);

    // ============ S3: energies e = v . tanh(q + m_proj + loc_proj(conv)) ============
    {
      const int gw = wg * 8 + wave;             // 128 waves, 4 row-tiles each (512 total)
      for (int rt = gw * 4; rt < gw * 4 + 4; ++rt) {
        const int row0 = rt * 16;
        const int b = row0 >> 8;                // 256 rows per batch element
        v16bf afr = load_a16(p.conv_out, OC_, row0, 0);   // 16 rows x 32 conv channels
        float eacc[8] = {0,0,0,0,0,0,0,0};
        for (int nt2 = 0; nt2 < 32; ++nt2) {
          int d = nt2 * 16 + nl;
          v16bf bw = load_b16(p.locp_wbf, OC_, nt2 * 16, 0);
          v8f c = {0.f,0.f,0.f,0.f,0.f,0.f,0.f,0.f};
          c = wmma_bf(afr, bw, c);              // one K=32 WMMA = full loc projection
          float qv = p.q[b * D_ + d];
          float lb = p.locp_b[d];
          float vw = p.v_w[d];
          #pragma unroll
          for (int v = 0; v < 8; ++v) {
            int row = row0 + v + 8 * half;
            float x = c[v] + lb + qv + p.mproj[(size_t)row * D_ + d];
            eacc[v] += tanh_(x) * vw;
          }
        }
        #pragma unroll
        for (int v = 0; v < 8; ++v) {           // reduce across 16 N-lanes (wave32 halves)
          float s = eacc[v];
          s += __shfl_xor(s, 1); s += __shfl_xor(s, 2);
          s += __shfl_xor(s, 4); s += __shfl_xor(s, 8);
          if (nl == 0) p.e[row0 + v + 8 * half] = s + p.v_b[0];
        }
      }
    }
    grid_barrier(p.bar, NWG);

    // ============ S4: softmax + context + next-step input assembly ============
    {
      const int b0 = wg * 2;
      if (wave == 0 || wave == 4) {             // one wave per batch row: softmax over 256
        int b = b0 + (wave >> 2);
        float ev[8], m = -1e30f;
        #pragma unroll
        for (int i = 0; i < 8; ++i) { ev[i] = p.e[b * TENC + lane + 32 * i]; m = fmaxf(m, ev[i]); }
        for (int off = 16; off; off >>= 1) m = fmaxf(m, __shfl_xor(m, off));
        float s = 0.f;
        #pragma unroll
        for (int i = 0; i < 8; ++i) { ev[i] = __expf(ev[i] - m); s += ev[i]; }
        for (int off = 16; off; off >>= 1) s += __shfl_xor(s, off);
        float inv = 1.f / s;
        #pragma unroll
        for (int i = 0; i < 8; ++i) {
          int tt = lane + 32 * i;
          float av = ev[i] * inv;
          p.a[b * TENC + tt] = av;
          p.acum[b * TENC + tt] += av;
        }
      }
      __syncthreads();
      {                                         // ctx[b,d] = sum_t a[b,t] * enc[b,t,d]
        int b = b0 + (wave >> 2);
        int d0 = (wave & 3) * 128 + lane * 4;
        float a0 = 0.f, a1 = 0.f, a2 = 0.f, a3 = 0.f;
        for (int tt = 0; tt < TENC; ++tt) {
          float av = p.a[b * TENC + tt];
          const float4 ev = *(const float4*)(p.enc + ((size_t)(b * TENC + tt)) * D_ + d0);
          a0 += av * ev.x; a1 += av * ev.y; a2 += av * ev.z; a3 += av * ev.w;
        }
        p.ctx[b * D_ + d0 + 0] = a0; p.ctx[b * D_ + d0 + 1] = a1;
        p.ctx[b * D_ + d0 + 2] = a2; p.ctx[b * D_ + d0 + 3] = a3;
        bf16_t* xa = p.xin_att + b * 768 + 256 + d0;      // att input = [pm(256) | ctx(512)]
        bf16_t* xd = p.xin_dec + b * 1024 + 512 + d0;     // dec input = [h_att(512) | ctx(512)]
        xa[0] = f2bf(a0); xa[1] = f2bf(a1); xa[2] = f2bf(a2); xa[3] = f2bf(a3);
        xd[0] = f2bf(a0); xd[1] = f2bf(a1); xd[2] = f2bf(a2); xd[3] = f2bf(a3);
      }
      if (t + 1 < TDEC) {                       // stage prenet output for next step
        for (int idx = tid; idx < 512; idx += 256) {
          int b = b0 + (idx >> 8), k = idx & 255;
          p.xin_att[b * 768 + k] = p.pm[((size_t)b * TDEC + (t + 1)) * 256 + k];
        }
      }
    }
    grid_barrier(p.bar, NWG);

    // ============ S5: decoder LSTM (K=1024 input, K=512 recurrent) ============
    {
      const int t0 = wave * 2;
      const int mi = t0 >> 3;
      const int n0 = (t0 & 7) * 16, n1 = n0 + 16;
      const int colbase = wg * 128;
      v8f c0 = {0.f,0.f,0.f,0.f,0.f,0.f,0.f,0.f};
      v8f c1 = {0.f,0.f,0.f,0.f,0.f,0.f,0.f,0.f};
      for (int k = 0; k < 1024; k += 32) {
        v16bf af = load_a16(p.xin_dec, 1024, mi * 16, k);
        v16bf b0 = load_b16(p.dec_wih, 1024, colbase + n0, k);
        v16bf b1 = load_b16(p.dec_wih, 1024, colbase + n1, k);
        c0 = wmma_bf(af, b0, c0);
        c1 = wmma_bf(af, b1, c1);
      }
      const bf16_t* hp = p.h_dec_bf + prev * B_ * D_;
      for (int k = 0; k < 512; k += 32) {
        v16bf af = load_a16(hp, 512, mi * 16, k);
        v16bf b0 = load_b16(p.dec_whh, 512, colbase + n0, k);
        v16bf b1 = load_b16(p.dec_whh, 512, colbase + n1, k);
        c0 = wmma_bf(af, b0, c0);
        c1 = wmma_bf(af, b1, c1);
      }
      #pragma unroll
      for (int v = 0; v < 8; ++v) {
        int b = mi * 16 + v + 8 * half;
        s_gates[b * 128 + n0 + nl] = c0[v];
        s_gates[b * 128 + n1 + nl] = c1[v];
      }
    }
    __syncthreads();
    for (int idx = tid; idx < B_ * 32; idx += 256) {
      int b = idx >> 5, jl = idx & 31;
      int col = wg * 128 + jl * 4;
      float gi = s_gates[b * 128 + jl * 4 + 0] + p.dec_b[col + 0];
      float gf = s_gates[b * 128 + jl * 4 + 1] + p.dec_b[col + 1];
      float gg = s_gates[b * 128 + jl * 4 + 2] + p.dec_b[col + 2];
      float go = s_gates[b * 128 + jl * 4 + 3] + p.dec_b[col + 3];
      int j = wg * 32 + jl;
      float cn = sigm_(gf) * p.c_dec[b * D_ + j] + sigm_(gi) * tanh_(gg);
      float h  = sigm_(go) * tanh_(cn);
      p.c_dec[b * D_ + j] = cn;
      p.h_dec[b * D_ + j] = h;
      p.h_dec_bf[cur * B_ * D_ + b * D_ + j] = f2bf(h);
    }
    grid_barrier(p.bar, NWG);

    // ============ S6: mel + stop heads ============
    {
      int b = gtid >> 7;            // 128 threads per batch row: 80 mel + 1 stop
      int j = gtid & 127;
      const float* h = p.h_dec + b * D_;
      if (j < MEL_) {
        const float* w = p.mel_w + (size_t)j * D_;
        float s = p.mel_b[j];
        for (int k = 0; k < D_; ++k) s += h[k] * w[k];
        p.out_mel[((size_t)b * TDEC + t) * MEL_ + j] = s;
      } else if (j == MEL_) {
        float s = p.stop_b[0];
        for (int k = 0; k < D_; ++k) s += h[k] * p.stop_w[k];
        p.out_stop[b * TDEC + t] = sigm_(s);
      }
    }
    // no barrier needed: next write of h_dec is 4 barriers away
  }
}

// =======================================================================
// Host launcher
// =======================================================================
extern "C" void kernel_launch(void* const* d_in, const int* in_sizes, int n_in,
                              void* d_out, int out_size, void* d_ws, size_t ws_size,
                              hipStream_t stream) {
  const float* enc     = (const float*)d_in[0];
  const float* mel_in  = (const float*)d_in[1];
  const float* pre_w1  = (const float*)d_in[2];
  const float* pre_b1  = (const float*)d_in[3];
  const float* pre_w2  = (const float*)d_in[4];
  const float* pre_b2  = (const float*)d_in[5];
  const float* att_wih = (const float*)d_in[6];
  const float* att_whh = (const float*)d_in[7];
  const float* att_bih = (const float*)d_in[8];
  const float* att_bhh = (const float*)d_in[9];
  const float* dec_wih = (const float*)d_in[10];
  const float* dec_whh = (const float*)d_in[11];
  const float* dec_bih = (const float*)d_in[12];
  const float* dec_bhh = (const float*)d_in[13];
  const float* q_w     = (const float*)d_in[14];
  const float* q_b     = (const float*)d_in[15];
  const float* mem_w   = (const float*)d_in[16];
  const float* mem_b   = (const float*)d_in[17];
  const float* conv_w  = (const float*)d_in[18];
  const float* conv_b  = (const float*)d_in[19];
  const float* locp_w  = (const float*)d_in[20];
  const float* locp_b  = (const float*)d_in[21];
  const float* v_w     = (const float*)d_in[22];
  const float* v_b     = (const float*)d_in[23];
  const float* mel_w   = (const float*)d_in[24];
  const float* mel_b   = (const float*)d_in[25];
  const float* stop_w  = (const float*)d_in[26];
  const float* stop_b  = (const float*)d_in[27];

  char* w = (char*)d_ws;
  size_t off = 0;
  auto alloc = [&](size_t bytes) -> void* {
    void* ptr = w + off;
    off = (off + bytes + 255) & ~(size_t)255;
    return ptr;
  };
  unsigned* bar       = (unsigned*)alloc(256);
  bf16_t* att_wih_bf  = (bf16_t*)alloc((size_t)2048 * 768 * 2);
  bf16_t* att_whh_bf  = (bf16_t*)alloc((size_t)2048 * 512 * 2);
  bf16_t* dec_wih_bf  = (bf16_t*)alloc((size_t)2048 * 1024 * 2);
  bf16_t* dec_whh_bf  = (bf16_t*)alloc((size_t)2048 * 512 * 2);
  bf16_t* q_w_bf      = (bf16_t*)alloc((size_t)512 * 512 * 2);
  bf16_t* mem_w_bf    = (bf16_t*)alloc((size_t)512 * 512 * 2);
  bf16_t* locp_w_bf   = (bf16_t*)alloc((size_t)512 * 32 * 2);
  bf16_t* pre_w2_bf   = (bf16_t*)alloc((size_t)256 * 256 * 2);
  float*  att_b_p     = (float*)alloc(2048 * 4);
  float*  dec_b_p     = (float*)alloc(2048 * 4);
  bf16_t* enc_bf      = (bf16_t*)alloc((size_t)B_ * TENC * D_ * 2);
  bf16_t* pm          = (bf16_t*)alloc((size_t)NROW * 256 * 2);
  bf16_t* h1          = (bf16_t*)alloc((size_t)NROW * 256 * 2);
  float*  mproj       = (float*)alloc((size_t)B_ * TENC * D_ * 4);
  float*  qbuf        = (float*)alloc((size_t)B_ * D_ * 4);
  float*  ebuf        = (float*)alloc((size_t)B_ * TENC * 4);
  float*  abuf        = (float*)alloc((size_t)B_ * TENC * 4);
  float*  acum        = (float*)alloc((size_t)B_ * TENC * 4);
  float*  ctx         = (float*)alloc((size_t)B_ * D_ * 4);
  float*  c_att       = (float*)alloc((size_t)B_ * D_ * 4);
  float*  c_dec       = (float*)alloc((size_t)B_ * D_ * 4);
  float*  h_dec       = (float*)alloc((size_t)B_ * D_ * 4);
  bf16_t* conv_out    = (bf16_t*)alloc((size_t)B_ * TENC * OC_ * 2);
  bf16_t* h_att_bf    = (bf16_t*)alloc((size_t)2 * B_ * D_ * 2);
  bf16_t* h_dec_bf    = (bf16_t*)alloc((size_t)2 * B_ * D_ * 2);
  bf16_t* xin_att     = (bf16_t*)alloc((size_t)B_ * 768 * 2);
  bf16_t* xin_dec     = (bf16_t*)alloc((size_t)B_ * 1024 * 2);
  (void)ws_size; (void)in_sizes; (void)n_in; (void)out_size;

  hipMemsetAsync(bar, 0, 256, stream);   // barrier state must start at 0 every call

  auto g = [](int n) { return (n + 255) / 256; };
  // weight packing (fp32 -> bf16, LSTM gate interleave)
  k_pack_plain<<<g(512 * 512), 256, 0, stream>>>(q_w, q_w_bf, 512 * 512);
  k_pack_plain<<<g(512 * 512), 256, 0, stream>>>(mem_w, mem_w_bf, 512 * 512);
  k_pack_plain<<<g(512 * 32), 256, 0, stream>>>(locp_w, locp_w_bf, 512 * 32);
  k_pack_plain<<<g(256 * 256), 256, 0, stream>>>(pre_w2, pre_w2_bf, 256 * 256);
  k_pack_plain<<<g(B_ * TENC * D_), 256, 0, stream>>>(enc, enc_bf, B_ * TENC * D_);
  k_pack_lstm_w<<<g(2048 * 768), 256, 0, stream>>>(att_wih, att_wih_bf, 768);
  k_pack_lstm_w<<<g(2048 * 512), 256, 0, stream>>>(att_whh, att_whh_bf, 512);
  k_pack_lstm_w<<<g(2048 * 1024), 256, 0, stream>>>(dec_wih, dec_wih_bf, 1024);
  k_pack_lstm_w<<<g(2048 * 512), 256, 0, stream>>>(dec_whh, dec_whh_bf, 512);
  k_pack_lstm_b<<<g(2048), 256, 0, stream>>>(att_bih, att_bhh, att_b_p);
  k_pack_lstm_b<<<g(2048), 256, 0, stream>>>(dec_bih, dec_bhh, dec_b_p);

  // hoisted pre-passes
  k_prenet1<<<NROW, 256, 0, stream>>>(mel_in, pre_w1, pre_b1, h1);
  k_prenet2<<<NROW * 16 / 8, 256, 0, stream>>>(h1, pre_w2_bf, pre_b2, pm);
  k_mproj<<<(B_ * TENC / 16) * 32 / 8, 256, 0, stream>>>(enc_bf, mem_w_bf, mem_b, mproj);

  // persistent decode
  DecP P;
  P.enc = enc; P.conv_w = conv_w; P.conv_b = conv_b; P.locp_b = locp_b;
  P.v_w = v_w; P.v_b = v_b; P.q_b = q_b; P.mel_w = mel_w; P.mel_b = mel_b;
  P.stop_w = stop_w; P.stop_b = stop_b;
  P.att_wih = att_wih_bf; P.att_whh = att_whh_bf;
  P.dec_wih = dec_wih_bf; P.dec_whh = dec_whh_bf;
  P.q_wbf = q_w_bf; P.locp_wbf = locp_w_bf; P.pm = pm;
  P.att_b = att_b_p; P.dec_b = dec_b_p; P.mproj = mproj;
  P.q = qbuf; P.e = ebuf; P.a = abuf; P.acum = acum; P.ctx = ctx;
  P.c_att = c_att; P.c_dec = c_dec; P.h_dec = h_dec;
  P.conv_out = conv_out; P.h_att_bf = h_att_bf; P.h_dec_bf = h_dec_bf;
  P.xin_att = xin_att; P.xin_dec = xin_dec;
  P.out_mel = (float*)d_out;
  P.out_stop = (float*)d_out + (size_t)B_ * TDEC * MEL_;
  P.bar = bar;
  k_decode<<<NWG, 256, 0, stream>>>(P);
}
